// HGAT_ESM2_V3_27470610825504
// MI455X (gfx1250) — compile-verified, matrix operand order checked by reading, not run on hardware
//
#include <hip/hip_runtime.h>

// ---------------------------------------------------------------------------
// HGAT-ESM2 forward for MI455X (gfx1250, wave32, WMMA bf16)
// Np=40000 Ng=20000 Dm=256 De=1280 H=512 C=1024 L=2
// Edges: pp=120000 pg=80000 gp=80000 gg=40000  (Etot=320000)
// TypeAttention's softmax is over per-dst-constant logits -> alpha = 1/indeg,
// so its GEMMs are dead code. Live GEMMs run on v_wmma_f32_16x16x32_bf16.
// ---------------------------------------------------------------------------

typedef __attribute__((ext_vector_type(16))) __bf16 v16bf;
typedef __attribute__((ext_vector_type(8)))  float  v8f;

#define HD 512
#define LSLOPE 0.2f

__device__ __forceinline__ unsigned short f2bfbits(float f) {
  unsigned u = __float_as_uint(f);
  u += 0x7FFFu + ((u >> 16) & 1u);          // round-to-nearest-even
  return (unsigned short)(u >> 16);
}
__device__ __forceinline__ __bf16 bfb(unsigned short h) {
  return __builtin_bit_cast(__bf16, h);
}
__device__ __forceinline__ unsigned fflip(float f) {   // order-preserving float->uint
  unsigned u = __float_as_uint(f);
  return (u & 0x80000000u) ? ~u : (u | 0x80000000u);
}
__device__ __forceinline__ float funflip(unsigned u) {
  return __uint_as_float((u & 0x80000000u) ? (u & 0x7FFFFFFFu) : ~u);
}

// ---- fp32 -> bf16-bits cast ------------------------------------------------
__global__ void k_cast_bf16(const float* __restrict__ s,
                            unsigned short* __restrict__ d, size_t n) {
  size_t i = (size_t)blockIdx.x * blockDim.x + threadIdx.x;
  if (i < n) d[i] = f2bfbits(s[i]);
}

__global__ void k_zero32(unsigned* __restrict__ p, size_t n) {
  size_t i = (size_t)blockIdx.x * blockDim.x + threadIdx.x;
  if (i < n) p[i] = 0u;
}

// ---- WMMA GEMM: out[r, c] = sum_k A[r,k]*W[c,k] (+bias[c]) ----------------
// A (M x K, lda): fp32 or bf16-bits (A_BF16). W bf16-bits (N x K row-major).
// out fp32 (ldo, coloff). Block = 128 threads = 4 waves; wave -> 16(M) x
// 128(N) tile (8 WMMA / A frag); grid.y * 128 = N. K % 32 == 0, M % 16 == 0.
template <bool A_BF16>
__global__ void k_gemm_wmma(const void* __restrict__ Av, int lda,
                            const unsigned short* __restrict__ W,
                            const float* __restrict__ bias,
                            float* __restrict__ out, int ldo, int coloff,
                            int M, int K) {
  const int lane = threadIdx.x & 31;
  const int wv   = threadIdx.x >> 5;
  const int mt   = blockIdx.x * 4 + wv;           // 16-row tile index
  if (mt * 16 >= M) return;                       // whole-wave uniform exit
  const int nb   = blockIdx.y * 128;
  const int rowA = mt * 16 + (lane & 15);
  const int ksub = (lane >> 4) * 8;               // 0 or 8

  v8f acc[8] = {v8f{}, v8f{}, v8f{}, v8f{}, v8f{}, v8f{}, v8f{}, v8f{}};
  for (int k = 0; k < K; k += 32) {
    v16bf af;
    if (A_BF16) {
      const unsigned short* pa =
          (const unsigned short*)Av + (size_t)rowA * lda + k + ksub;
      #pragma unroll
      for (int i = 0; i < 8; ++i) af[i]     = bfb(pa[i]);
      #pragma unroll
      for (int i = 0; i < 8; ++i) af[8 + i] = bfb(pa[16 + i]);
      if (k + 32 < K) __builtin_prefetch(pa + 32, 0, 1);   // global_prefetch_b8
    } else {
      const float* pa = (const float*)Av + (size_t)rowA * lda + k + ksub;
      #pragma unroll
      for (int i = 0; i < 8; ++i) af[i]     = bfb(f2bfbits(pa[i]));
      #pragma unroll
      for (int i = 0; i < 8; ++i) af[8 + i] = bfb(f2bfbits(pa[16 + i]));
      if (k + 32 < K) __builtin_prefetch(pa + 32, 0, 1);
    }
    #pragma unroll
    for (int n = 0; n < 8; ++n) {
      const unsigned short* pw =
          W + (size_t)(nb + n * 16 + (lane & 15)) * K + k + ksub;
      v16bf bfr;
      #pragma unroll
      for (int i = 0; i < 8; ++i) bfr[i]     = bfb(pw[i]);
      #pragma unroll
      for (int i = 0; i < 8; ++i) bfr[8 + i] = bfb(pw[16 + i]);
      acc[n] = __builtin_amdgcn_wmma_f32_16x16x32_bf16(
          false, af, false, bfr, (short)0, acc[n], false, false);
    }
  }
  const int rbase = mt * 16 + ksub;   // C/D layout: rows i + (lane>=16 ? 8 : 0)
  const int c0    = lane & 15;
  #pragma unroll
  for (int n = 0; n < 8; ++n) {
    const int col = nb + n * 16 + c0;
    const float bv = bias ? bias[col] : 0.0f;
    #pragma unroll
    for (int i = 0; i < 8; ++i)
      out[(size_t)(rbase + i) * ldo + coloff + col] = acc[n][i] + bv;
  }
}

// ---- row LayerNorm over H=512, one wave per row ---------------------------
// Writes fp32 (outf, nullable) and/or bf16-bits (outb, nullable).
__global__ void k_layernorm(const float* __restrict__ x, const float* __restrict__ g,
                            const float* __restrict__ b,
                            float* __restrict__ outf,
                            unsigned short* __restrict__ outb) {
  const int row = blockIdx.x;
  const int lane = threadIdx.x;
  const float* xr = x + (size_t)row * HD;
  float v[16];
  float s = 0.f;
  #pragma unroll
  for (int i = 0; i < 16; ++i) { v[i] = xr[lane + i * 32]; s += v[i]; }
  #pragma unroll
  for (int m = 16; m >= 1; m >>= 1) s += __shfl_xor(s, m, 32);
  const float mu = s * (1.0f / HD);
  float q = 0.f;
  #pragma unroll
  for (int i = 0; i < 16; ++i) { float d = v[i] - mu; q += d * d; }
  #pragma unroll
  for (int m = 16; m >= 1; m >>= 1) q += __shfl_xor(q, m, 32);
  const float inv = rsqrtf(q * (1.0f / HD) + 1e-5f);
  #pragma unroll
  for (int i = 0; i < 16; ++i) {
    int c = lane + i * 32;
    float o = (v[i] - mu) * inv * g[c] + b[c];
    if (outf) outf[(size_t)row * HD + c] = o;
    if (outb) outb[(size_t)row * HD + c] = f2bfbits(o);
  }
}

// ---- edge assembly: homogeneous ids + per-relation in-degree counts -------
__global__ void k_build_edges(const int* __restrict__ s, const int* __restrict__ d,
                              int E, int soff, int doff,
                              int* __restrict__ sh, int* __restrict__ dh,
                              int* __restrict__ cnt) {
  int i = blockIdx.x * blockDim.x + threadIdx.x;
  if (i >= E) return;
  int dv = d[i];
  sh[i] = s[i] + soff;
  dh[i] = dv + doff;
  atomicAdd(&cnt[dv], 1);
}

// alpha_e = 1/indeg_rel(dst): exact value of the reference's per-relation
// edge_softmax of a per-dst-constant logit.
__global__ void k_alpha(const int* __restrict__ d, const int* __restrict__ cnt,
                        float* __restrict__ alpha, int E) {
  int i = blockIdx.x * blockDim.x + threadIdx.x;
  if (i >= E) return;
  alpha[i] = 1.0f / (float)cnt[d[i]];
}

// ---- e_edge = sum_j leaky((hl[src,j]+hr[dst,j])*alpha); wave per edge -----
// hl/hr are column slices of the fused hlr buffer (ld = 1024).
__global__ void k_edge_score(const int* __restrict__ sh, const int* __restrict__ dh,
                             const float* __restrict__ alpha,
                             const float* __restrict__ hl, const float* __restrict__ hr,
                             int ldh, float* __restrict__ eo, int E) {
  int w = (int)(((size_t)blockIdx.x * blockDim.x + threadIdx.x) >> 5);
  int lane = threadIdx.x & 31;
  if (w >= E) return;
  const int s = sh[w], d = dh[w];
  const float a = alpha[w];
  const float* pl = hl + (size_t)s * ldh;
  const float* pr = hr + (size_t)d * ldh;
  float sum = 0.f;
  #pragma unroll
  for (int i = 0; i < 16; ++i) {
    int c = lane + i * 32;
    float z = (pl[c] + pr[c]) * a;
    sum += (z > 0.f) ? z : LSLOPE * z;
  }
  #pragma unroll
  for (int m = 16; m >= 1; m >>= 1) sum += __shfl_xor(sum, m, 32);
  if (lane == 0) eo[w] = sum;
}

// ---- segment softmax over dst (N_hom groups) ------------------------------
__global__ void k_init_max(unsigned* __restrict__ m, int n) {
  int i = blockIdx.x * blockDim.x + threadIdx.x;
  if (i < n) m[i] = fflip(-3.0e38f);
}
__global__ void k_edge_max(const int* __restrict__ dh, const float* __restrict__ e,
                           unsigned* __restrict__ m, int E) {
  int i = blockIdx.x * blockDim.x + threadIdx.x;
  if (i >= E) return;
  atomicMax(&m[dh[i]], fflip(e[i]));
}
__global__ void k_edge_expsum(const int* __restrict__ dh, const float* __restrict__ e,
                              const unsigned* __restrict__ m,
                              float* __restrict__ ex, float* __restrict__ ss, int E) {
  int i = blockIdx.x * blockDim.x + threadIdx.x;
  if (i >= E) return;
  int d = dh[i];
  float v = expf(e[i] - funflip(m[d]));
  ex[i] = v;
  atomicAdd(&ss[d], v);
}

// ---- xo[dst] += (ex/ssum) * x[src]; wave per edge -------------------------
__global__ void k_scatter(const int* __restrict__ sh, const int* __restrict__ dh,
                          const float* __restrict__ ex, const float* __restrict__ ss,
                          const float* __restrict__ x, float* __restrict__ xo, int E) {
  int w = (int)(((size_t)blockIdx.x * blockDim.x + threadIdx.x) >> 5);
  int lane = threadIdx.x & 31;
  if (w >= E) return;
  const int s = sh[w], d = dh[w];
  const float a = ex[w] / ss[d];
  const float* xr = x + (size_t)s * HD;
  float* xow = xo + (size_t)d * HD;
  #pragma unroll
  for (int i = 0; i < 16; ++i) {
    int c = lane + i * 32;
    atomicAdd(&xow[c], a * xr[c]);
  }
}

// ===========================================================================
extern "C" void kernel_launch(void* const* d_in, const int* in_sizes, int n_in,
                              void* d_out, int out_size, void* d_ws, size_t ws_size,
                              hipStream_t stream) {
  (void)in_sizes; (void)n_in; (void)out_size; (void)ws_size;
  const int Np = 40000, Ng = 20000, Dm = 256, De = 1280, H = 512, C = 1024;
  const int Nhom = Np + Ng;
  const int Epp = 120000, Epg = 80000, Egp = 80000, Egg = 40000;
  const int Etot = Epp + Epg + Egp + Egg;

  const float* xp_msa = (const float*)d_in[0];
  const float* xg_msa = (const float*)d_in[1];
  const float* xp_esm = (const float*)d_in[2];
  const float* xg_esm = (const float*)d_in[3];
  const float* Wmsa_p = (const float*)d_in[4];
  const float* bmsa_p = (const float*)d_in[5];
  const float* Wmsa_g = (const float*)d_in[6];
  const float* bmsa_g = (const float*)d_in[7];
  const float* Wesm_p = (const float*)d_in[8];
  const float* Wesm_g = (const float*)d_in[9];
  const float* lnf_p_g = (const float*)d_in[10];
  const float* lnf_p_b = (const float*)d_in[11];
  const float* lnf_g_g = (const float*)d_in[12];
  const float* lnf_g_b = (const float*)d_in[13];
  // d_in[14..18]: mul_W/mul_b/mur_W/mur_b/attn_w -> provably dead (uniform softmax)
  const float* node_Wl = (const float*)d_in[19];
  const float* node_Wr = (const float*)d_in[20];
  const float* head_g  = (const float*)d_in[21];
  const float* head_b  = (const float*)d_in[22];
  const float* Wcls    = (const float*)d_in[23];
  const float* bcls    = (const float*)d_in[24];
  const int* pp_src = (const int*)d_in[25];
  const int* pp_dst = (const int*)d_in[26];
  const int* pg_src = (const int*)d_in[27];
  const int* pg_dst = (const int*)d_in[28];
  const int* gp_src = (const int*)d_in[29];
  const int* gp_dst = (const int*)d_in[30];
  const int* gg_src = (const int*)d_in[31];
  const int* gg_dst = (const int*)d_in[32];

  // ---- workspace layout ----
  char* ws = (char*)d_ws;
  size_t off = 0;
  auto alloc = [&](size_t bytes) -> char* {
    char* p = ws + off; off = (off + bytes + 255) & ~(size_t)255; return p;
  };
  float* h0  = (float*)alloc((size_t)Nhom * H * 4);
  float* h1  = (float*)alloc((size_t)Nhom * H * 4);
  float* hlr = (float*)alloc((size_t)Nhom * 2 * H * 4);   // [hl | hr], ld = 1024
  unsigned short* hb = (unsigned short*)alloc((size_t)Nhom * H * 2);  // bf16 A
  unsigned short* wb_msa_p = (unsigned short*)alloc((size_t)256 * Dm * 2);
  unsigned short* wb_msa_g = (unsigned short*)alloc((size_t)256 * Dm * 2);
  unsigned short* wb_esm_p = (unsigned short*)alloc((size_t)256 * De * 2);
  unsigned short* wb_esm_g = (unsigned short*)alloc((size_t)256 * De * 2);
  unsigned short* wb_node  = (unsigned short*)alloc((size_t)2 * 2 * H * H * 2);
  unsigned short* wb_cls   = (unsigned short*)alloc((size_t)C * H * 2);
  int*   src_h = (int*)alloc((size_t)Etot * 4);
  int*   dst_h = (int*)alloc((size_t)Etot * 4);
  float* alpha = (float*)alloc((size_t)Etot * 4);
  float* esc   = (float*)alloc((size_t)Etot * 4);
  float* exb   = (float*)alloc((size_t)Etot * 4);
  int*   cnt   = (int*)alloc((size_t)(Np + Ng + Np + Ng) * 4);
  unsigned* mmax = (unsigned*)alloc((size_t)Nhom * 4);
  float*    ssum = (float*)alloc((size_t)Nhom * 4);

  auto grid1 = [](size_t n, int b) { return dim3((unsigned)((n + b - 1) / b)); };
  auto cast = [&](const float* s, unsigned short* d, size_t n) {
    k_cast_bf16<<<grid1(n, 256), dim3(256), 0, stream>>>(s, d, n);
  };
  auto gemm_f32a = [&](const float* A, int lda, const unsigned short* W,
                       const float* bias, float* out, int ldo, int coloff,
                       int M, int N, int K) {
    dim3 g((unsigned)((M / 16 + 3) / 4), (unsigned)(N / 128));
    k_gemm_wmma<false><<<g, dim3(128), 0, stream>>>(A, lda, W, bias, out, ldo,
                                                    coloff, M, K);
  };
  auto gemm_bf16a = [&](const unsigned short* A, int lda, const unsigned short* W,
                        const float* bias, float* out, int ldo, int coloff,
                        int M, int N, int K) {
    dim3 g((unsigned)((M / 16 + 3) / 4), (unsigned)(N / 128));
    k_gemm_wmma<true><<<g, dim3(128), 0, stream>>>(A, lda, W, bias, out, ldo,
                                                   coloff, M, K);
  };

  // ---- 0) weight casts to bf16; node weights fused [Wl_l ; Wr_l] ----------
  cast(Wmsa_p, wb_msa_p, (size_t)256 * Dm);
  cast(Wmsa_g, wb_msa_g, (size_t)256 * Dm);
  cast(Wesm_p, wb_esm_p, (size_t)256 * De);
  cast(Wesm_g, wb_esm_g, (size_t)256 * De);
  for (int l = 0; l < 2; ++l) {
    cast(node_Wl + (size_t)l * H * H, wb_node + (size_t)l * 2 * H * H, (size_t)H * H);
    cast(node_Wr + (size_t)l * H * H, wb_node + (size_t)l * 2 * H * H + (size_t)H * H,
         (size_t)H * H);
  }
  cast(Wcls, wb_cls, (size_t)C * H);

  // ---- 1) fused input projections -> h0 (pre-LN); LN -> h0 (f32) + hb (bf16)
  gemm_f32a(xp_msa, Dm, wb_msa_p, bmsa_p, h0,                  H, 0,   Np, 256, Dm);
  gemm_f32a(xp_esm, De, wb_esm_p, nullptr, h0,                 H, 256, Np, 256, De);
  gemm_f32a(xg_msa, Dm, wb_msa_g, bmsa_g, h0 + (size_t)Np * H, H, 0,   Ng, 256, Dm);
  gemm_f32a(xg_esm, De, wb_esm_g, nullptr, h0 + (size_t)Np * H, H, 256, Ng, 256, De);
  k_layernorm<<<dim3(Np), dim3(32), 0, stream>>>(h0, lnf_p_g, lnf_p_b, h0, hb);
  k_layernorm<<<dim3(Ng), dim3(32), 0, stream>>>(h0 + (size_t)Np * H, lnf_g_g, lnf_g_b,
                                                 h0 + (size_t)Np * H,
                                                 hb + (size_t)Np * H);

  // ---- 2) graph structure: homogeneous edge ids + alpha = 1/indeg_rel(dst) ----
  k_zero32<<<grid1(Np + Ng + Np + Ng, 256), dim3(256), 0, stream>>>(
      (unsigned*)cnt, (size_t)(Np + Ng + Np + Ng));
  k_build_edges<<<grid1(Epp, 256), dim3(256), 0, stream>>>(pp_src, pp_dst, Epp, 0, 0,
      src_h, dst_h, cnt);
  k_build_edges<<<grid1(Epg, 256), dim3(256), 0, stream>>>(pg_src, pg_dst, Epg, 0, Np,
      src_h + Epp, dst_h + Epp, cnt + Np);
  k_build_edges<<<grid1(Egp, 256), dim3(256), 0, stream>>>(gp_src, gp_dst, Egp, Np, 0,
      src_h + Epp + Epg, dst_h + Epp + Epg, cnt + Np + Ng);
  k_build_edges<<<grid1(Egg, 256), dim3(256), 0, stream>>>(gg_src, gg_dst, Egg, Np, Np,
      src_h + Epp + Epg + Egp, dst_h + Epp + Epg + Egp, cnt + Np + Ng + Np);
  k_alpha<<<grid1(Epp, 256), dim3(256), 0, stream>>>(pp_dst, cnt, alpha, Epp);
  k_alpha<<<grid1(Epg, 256), dim3(256), 0, stream>>>(pg_dst, cnt + Np, alpha + Epp, Epg);
  k_alpha<<<grid1(Egp, 256), dim3(256), 0, stream>>>(gp_dst, cnt + Np + Ng,
      alpha + Epp + Epg, Egp);
  k_alpha<<<grid1(Egg, 256), dim3(256), 0, stream>>>(gg_dst, cnt + Np + Ng + Np,
      alpha + Epp + Epg + Egp, Egg);

  // ---- 3) layers ----
  for (int l = 0; l < 2; ++l) {
    float* cur = (l == 0) ? h0 : h1;
    float* nxt = (l == 0) ? h1 : h0;
    const unsigned short* Wn = wb_node + (size_t)l * 2 * H * H;
    if (l > 0)  // layer 0's hb comes straight from the LayerNorm
      cast(cur, hb, (size_t)Nhom * H);
    // fused hl|hr: one M=60000, N=1024, K=512 WMMA GEMM, bf16 A
    gemm_bf16a(hb, H, Wn, nullptr, hlr, 2 * H, 0, Nhom, 2 * H, H);
    k_edge_score<<<grid1((size_t)Etot * 32, 256), dim3(256), 0, stream>>>(
        src_h, dst_h, alpha, hlr, hlr + H, 2 * H, esc, Etot);
    k_init_max<<<grid1(Nhom, 256), dim3(256), 0, stream>>>(mmax, Nhom);
    k_edge_max<<<grid1(Etot, 256), dim3(256), 0, stream>>>(dst_h, esc, mmax, Etot);
    k_zero32<<<grid1(Nhom, 256), dim3(256), 0, stream>>>((unsigned*)ssum, (size_t)Nhom);
    k_edge_expsum<<<grid1(Etot, 256), dim3(256), 0, stream>>>(dst_h, esc, mmax, exb,
                                                              ssum, Etot);
    k_zero32<<<grid1((size_t)Nhom * H, 256), dim3(256), 0, stream>>>(
        (unsigned*)nxt, (size_t)Nhom * H);
    k_scatter<<<grid1((size_t)Etot * 32, 256), dim3(256), 0, stream>>>(
        src_h, dst_h, exb, ssum, cur, nxt, Etot);
  }

  // ---- 4) head: LayerNorm(h[0]) -> bf16, then @ Wcls^T + bcls -> d_out ----
  k_layernorm<<<dim3(Np), dim3(32), 0, stream>>>(h0, head_g, head_b, nullptr, hb);
  gemm_bf16a(hb, H, wb_cls, bcls, (float*)d_out, C, 0, Np, C, H);
}
